// WrappedMinGRU_20280835572037
// MI455X (gfx1250) — compile-verified, matrix operand order checked by reading, not run on hardware
//
#include <hip/hip_runtime.h>
#include <hip/hip_bf16.h>

typedef __attribute__((ext_vector_type(16))) __bf16 v16bf;
typedef __attribute__((ext_vector_type(8)))  float  v8f;
typedef __attribute__((ext_vector_type(4)))  int    v4i;

// typed pointers matching __builtin_amdgcn_global_load_async_to_lds_b128 params
typedef __attribute__((address_space(1))) v4i* g_v4i_p;   // global (device) src
typedef __attribute__((address_space(3))) v4i* l_v4i_p;   // LDS dst

#define B_ 4
#define S_ 4096
#define D_ 1024
#define H_ 1024
#define M_ (B_ * S_)   // 16384 rows into the GEMMs
#define NC_ 64         // chunks per sequence
#define LC_ 64         // steps per chunk (NC_*LC_ == S_)

#define GEMM_K 1024
#define GEMM_N 1024
#define TILE_N 64                                  // N columns per block
#define LDS_TILE_SHORTS (TILE_N * GEMM_K)          // 64K bf16 = 128 KB per matrix
#define LDS_TOTAL_BYTES (2 * LDS_TILE_SHORTS * 2)  // 256 KB (both matrices)
#define KSTEPS (GEMM_K / 32)                       // 32 WMMA k-steps

// ---------------- helpers ----------------

__device__ __forceinline__ unsigned short f2bf(float f) {
    union { float f; unsigned u; } x; x.f = f;
    unsigned u = x.u + 0x7FFFu + ((x.u >> 16) & 1u);  // round-to-nearest-even
    return (unsigned short)(u >> 16);
}

__device__ __forceinline__ float sigmoidf_(float x) {
    return 1.0f / (1.0f + __expf(-x));
}

__device__ __forceinline__ v8f zero8() {
    v8f v;
#pragma unroll
    for (int i = 0; i < 8; ++i) v[i] = 0.0f;
    return v;
}

union bfrag { uint4 q[2]; v16bf v; };

// ---------------- fp32 -> bf16 conversion ----------------

__global__ void cvt_f32_bf16_kernel(const float* __restrict__ src,
                                    unsigned short* __restrict__ dst, int n) {
    int i = blockIdx.x * blockDim.x + threadIdx.x;
    int stride = gridDim.x * blockDim.x;
    for (; i < n; i += stride) dst[i] = f2bf(src[i]);
}

// ---------------- fused dual-gate GEMM (WMMA bf16, LDS-staged weights) ----
// A: (M_, K) bf16 row-major.  Wz/Wh: (N, K) bf16 row-major (N=K=1024).
// Computes kz = A@Wz^T + bz, ph = A@Wh^T + bh, then stores
//   Zout = sigmoid(kz),  Tout = g(ph) = (ph>=0 ? ph+0.5 : sigmoid(ph))
// Block: 256 threads (8 waves). Wave tile: 16(M) x 64(N). Block tile: 128 x 64.
// Weight tile is staged into LDS *pre-swizzled into WMMA fragment order* via
// async global->LDS copies (linear global reads, scattered LDS writes).
// LDS layout: chunk (i, f, lane, c) at byte ((i*8+f)*32 + lane)*32 + c*16,
//   i = k-step (k = i*32), f = mat*4 + ntile, c = 16B half of the 32B fragment.
__global__ void gate_gemm_kernel(const unsigned short* __restrict__ A,
                                 const unsigned short* __restrict__ Wz,
                                 const unsigned short* __restrict__ Wh,
                                 const float* __restrict__ bz,
                                 const float* __restrict__ bh,
                                 float* __restrict__ Zout,
                                 float* __restrict__ Tout) {
    const int K = GEMM_K, N = GEMM_N;
    const int lane  = threadIdx.x & 31;
    const int wave  = threadIdx.x >> 5;
    const int m0    = blockIdx.x * 128 + wave * 16;
    const int n0    = blockIdx.y * TILE_N;
    const int lrow  = lane & 15;          // row (A) / col (B) within 16-tile
    const int khalf = (lane >> 4) * 8;    // lanes 16-31 handle K+8 chunk

    extern __shared__ unsigned short lds_w[];   // swizzled, 256 KB

    // ---- stage + swizzle weight tile into LDS (async, linear global reads) --
    const unsigned short* gz = Wz + (size_t)n0 * K;
    const unsigned short* gh = Wh + (size_t)n0 * K;
    // 8192 16B-chunks per matrix, 16384 total
    for (int g = threadIdx.x; g < 2 * 8192; g += 256) {
        int mat  = g >> 13;
        int gg   = g & 8191;
        int nrow = gg >> 7;          // row within tile (0..63)
        int kc   = gg & 127;         // 16B chunk within row (0..127)
        int i    = kc >> 2;          // k-step
        int kh   = kc & 1;           // which lane half
        int c    = (kc >> 1) & 1;    // which 16B half of fragment
        int f    = mat * 4 + (nrow >> 4);
        int lf   = kh * 16 + (nrow & 15);
        unsigned lds_off = (((unsigned)(i * 8 + f) * 32 + lf) * 32 + c * 16);  // bytes
        const unsigned short* src = (mat ? gh : gz) + (size_t)gg * 8;
#if __has_builtin(__builtin_amdgcn_global_load_async_to_lds_b128)
        __builtin_amdgcn_global_load_async_to_lds_b128(
            (g_v4i_p)src, (l_v4i_p)((char*)lds_w + lds_off), 0, 0);
#else
        *(uint4*)((char*)lds_w + lds_off) = *(const uint4*)src;
#endif
    }
#if __has_builtin(__builtin_amdgcn_global_load_async_to_lds_b128)
#if __has_builtin(__builtin_amdgcn_s_wait_asynccnt)
    __builtin_amdgcn_s_wait_asynccnt(0);
#else
    asm volatile("s_wait_asynccnt 0x0" ::: "memory");
#endif
#endif
    __syncthreads();

    v8f cz[4], ch[4];
#pragma unroll
    for (int t = 0; t < 4; ++t) { cz[t] = zero8(); ch[t] = zero8(); }

    const uint4* arow = (const uint4*)(A + (size_t)(m0 + lrow) * K);
    // per-lane base into swizzled LDS: fragment f of step i at +(i*8+f)*1024B
    const unsigned short* lbase = lds_w + (size_t)lane * 16;

#pragma unroll 2
    for (int i = 0; i < KSTEPS; ++i) {
        const int e0 = i * 32 + khalf;
        bfrag af;
        af.q[0] = arow[e0 >> 3];
        af.q[1] = arow[(e0 + 16) >> 3];
        // batch all 8 B-fragments, then fire 8 WMMAs back-to-back
        bfrag bzf[4], bhf[4];
#pragma unroll
        for (int t = 0; t < 4; ++t) {
            const uint4* pz = (const uint4*)(lbase + (size_t)(i * 8 + t) * 512);
            const uint4* ph = (const uint4*)(lbase + (size_t)(i * 8 + 4 + t) * 512);
            bzf[t].q[0] = pz[0]; bzf[t].q[1] = pz[1];
            bhf[t].q[0] = ph[0]; bhf[t].q[1] = ph[1];
        }
#pragma unroll
        for (int t = 0; t < 4; ++t)
            cz[t] = __builtin_amdgcn_wmma_f32_16x16x32_bf16(
                false, af.v, false, bzf[t].v, (short)0, cz[t], false, false);
#pragma unroll
        for (int t = 0; t < 4; ++t)
            ch[t] = __builtin_amdgcn_wmma_f32_16x16x32_bf16(
                false, af.v, false, bhf[t].v, (short)0, ch[t], false, false);
    }

    // Epilogue: C layout -> VGPR r holds (M = r + 8*(lane>=16), N = lane&15)
    const int rbump = (lane >> 4) << 3;
#pragma unroll
    for (int t = 0; t < 4; ++t) {
        int col = n0 + t * 16 + lrow;
        float bzv = bz[col];
        float bhv = bh[col];
#pragma unroll
        for (int r = 0; r < 8; ++r) {
            int row = m0 + r + rbump;
            size_t idx = (size_t)row * N + col;
            float kz = cz[t][r] + bzv;
            float ph = ch[t][r] + bhv;
            float z  = sigmoidf_(kz);
            float ht = (ph >= 0.0f) ? (ph + 0.5f) : sigmoidf_(ph);
            Zout[idx] = z;
            Tout[idx] = ht;
        }
    }
}

// ---------------- chunked affine scan ----------------
// Recurrence: h[s] = c[s]*h[s-1] + v[s],  c = 1-z, v = z*h_tilde, h[-1] = 0.5.

// Pass 1: per (b, chunk j, channel hc): A_c = prod(c), B_c = scan from 0.
__global__ void scan_pass1_kernel(const float* __restrict__ Z,
                                  const float* __restrict__ T,
                                  float* __restrict__ Ac,
                                  float* __restrict__ Bc) {
    int tid = blockIdx.x * blockDim.x + threadIdx.x;   // < B_*NC_*H_
    int hc = tid & (H_ - 1);
    int j  = (tid >> 10) & (NC_ - 1);
    int b  = tid >> 16;
    size_t base = ((size_t)b * S_ + (size_t)j * LC_) * H_ + hc;
    float Aacc = 1.0f, acc = 0.0f;
    for (int s = 0; s < LC_; ++s) {
        float z = Z[base + (size_t)s * H_];
        float t = T[base + (size_t)s * H_];
        float c = 1.0f - z;
        Aacc *= c;
        acc = c * acc + z * t;
    }
    Ac[tid] = Aacc;
    Bc[tid] = acc;
}

// Pass 2: per (b, hc): scan over NC_ chunks to get each chunk's initial h.
__global__ void scan_pass2_kernel(const float* __restrict__ Ac,
                                  const float* __restrict__ Bc,
                                  float* __restrict__ P) {
    int tid = blockIdx.x * blockDim.x + threadIdx.x;   // < B_*H_
    int hc = tid & (H_ - 1);
    int b  = tid >> 10;
    float p = 0.5f;                                    // h init = g(0) = 0.5
    for (int j = 0; j < NC_; ++j) {
        size_t ci = ((size_t)b * NC_ + j) * H_ + hc;
        P[ci] = p;
        p = Ac[ci] * p + Bc[ci];
    }
}

// Pass 3: replay each chunk from its true initial value, write outputs.
// MODE 0: write bf16 hidden states (feeds next layer's GEMM).
// MODE 1: write f32 hidden states (final h2 into d_out).
template <int MODE>
__global__ void scan_pass3_kernel(const float* __restrict__ Z,
                                  const float* __restrict__ T,
                                  const float* __restrict__ P,
                                  unsigned short* __restrict__ out_bf16,
                                  float* __restrict__ out_f32,
                                  float* __restrict__ hn) {               // (B_,H_) last step
    int tid = blockIdx.x * blockDim.x + threadIdx.x;   // < B_*NC_*H_
    int hc = tid & (H_ - 1);
    int j  = (tid >> 10) & (NC_ - 1);
    int b  = tid >> 16;
    size_t base = ((size_t)b * S_ + (size_t)j * LC_) * H_ + hc;
    float hval = P[((size_t)b * NC_ + j) * H_ + hc];
    for (int s = 0; s < LC_; ++s) {
        size_t gidx = base + (size_t)s * H_;
        float z = Z[gidx];
        float t = T[gidx];
        hval = hval + z * (t - hval);
        if (MODE == 0) out_bf16[gidx] = f2bf(hval);
        else           out_f32[gidx]  = hval;
    }
    if (j == NC_ - 1) hn[(size_t)b * H_ + hc] = hval;
}

// ---------------- launch ----------------

extern "C" void kernel_launch(void* const* d_in, const int* in_sizes, int n_in,
                              void* d_out, int out_size, void* d_ws, size_t ws_size,
                              hipStream_t stream) {
    (void)in_sizes; (void)n_in; (void)out_size; (void)ws_size;

    const float* x   = (const float*)d_in[0];
    const float* Wz0 = (const float*)d_in[1];
    const float* bz0 = (const float*)d_in[2];
    const float* Wh0 = (const float*)d_in[3];
    const float* bh0 = (const float*)d_in[4];
    const float* Wz1 = (const float*)d_in[5];
    const float* bz1 = (const float*)d_in[6];
    const float* Wh1 = (const float*)d_in[7];
    const float* bh1 = (const float*)d_in[8];

    float* out = (float*)d_out;                 // h2: (B,S,H) then h_n: (2,B,H)
    float* hn0 = out + (size_t)B_ * S_ * H_;
    float* hn1 = hn0 + (size_t)B_ * H_;

    size_t off = 0;
    auto wsp = [&](size_t bytes) {
        void* p = (char*)d_ws + off;
        off += (bytes + 255) & ~(size_t)255;
        return p;
    };
    unsigned short* xb   = (unsigned short*)wsp((size_t)M_ * D_ * 2);   // 32 MB
    unsigned short* h1b  = (unsigned short*)wsp((size_t)M_ * H_ * 2);   // 32 MB
    unsigned short* wz0b = (unsigned short*)wsp((size_t)H_ * D_ * 2);   // 2 MB x4
    unsigned short* wh0b = (unsigned short*)wsp((size_t)H_ * D_ * 2);
    unsigned short* wz1b = (unsigned short*)wsp((size_t)H_ * H_ * 2);
    unsigned short* wh1b = (unsigned short*)wsp((size_t)H_ * H_ * 2);
    float* Zbuf = (float*)wsp((size_t)M_ * H_ * 4);                     // 64 MB
    float* Tbuf = (float*)wsp((size_t)M_ * H_ * 4);                     // 64 MB
    float* Ac   = (float*)wsp((size_t)B_ * NC_ * H_ * 4);               // 1 MB x3
    float* Bc   = (float*)wsp((size_t)B_ * NC_ * H_ * 4);
    float* P    = (float*)wsp((size_t)B_ * NC_ * H_ * 4);

    const int cvtThreads = 256;
    // Convert inputs/weights to bf16
    cvt_f32_bf16_kernel<<<1024, cvtThreads, 0, stream>>>(x, xb, M_ * D_);
    cvt_f32_bf16_kernel<<<256, cvtThreads, 0, stream>>>(Wz0, wz0b, H_ * D_);
    cvt_f32_bf16_kernel<<<256, cvtThreads, 0, stream>>>(Wh0, wh0b, H_ * D_);
    cvt_f32_bf16_kernel<<<256, cvtThreads, 0, stream>>>(Wz1, wz1b, H_ * H_);
    cvt_f32_bf16_kernel<<<256, cvtThreads, 0, stream>>>(Wh1, wh1b, H_ * H_);

    dim3 gemmGrid(M_ / 128, GEMM_N / TILE_N);   // (128, 16)
    dim3 gemmBlock(256);
    const int p1Blocks = (B_ * NC_ * H_) / 256;   // 1024
    const int p2Blocks = (B_ * H_) / 256;         // 16

    // ---- Layer 0 ----
    gate_gemm_kernel<<<gemmGrid, gemmBlock, LDS_TOTAL_BYTES, stream>>>(
        xb, wz0b, wh0b, bz0, bh0, Zbuf, Tbuf);
    scan_pass1_kernel<<<p1Blocks, 256, 0, stream>>>(Zbuf, Tbuf, Ac, Bc);
    scan_pass2_kernel<<<p2Blocks, 256, 0, stream>>>(Ac, Bc, P);
    scan_pass3_kernel<0><<<p1Blocks, 256, 0, stream>>>(Zbuf, Tbuf, P, h1b, nullptr, hn0);

    // ---- Layer 1 ----
    gate_gemm_kernel<<<gemmGrid, gemmBlock, LDS_TOTAL_BYTES, stream>>>(
        h1b, wz1b, wh1b, bz1, bh1, Zbuf, Tbuf);
    scan_pass1_kernel<<<p1Blocks, 256, 0, stream>>>(Zbuf, Tbuf, Ac, Bc);
    scan_pass2_kernel<<<p2Blocks, 256, 0, stream>>>(Ac, Bc, P);
    scan_pass3_kernel<1><<<p1Blocks, 256, 0, stream>>>(Zbuf, Tbuf, P, nullptr, out, hn1);
}
